// Simulator_13099650253459
// MI455X (gfx1250) — compile-verified, matrix operand order checked by reading, not run on hardware
//
#include <hip/hip_runtime.h>
#include <hip/hip_bf16.h>

typedef __attribute__((ext_vector_type(16))) _Float16 v16h;
typedef __attribute__((ext_vector_type(8)))  float    v8f;

#define N_SEQ   64
#define N_WIN   4077
#define W_WIN   12
#define FEAT    32
#define HID     64
#define SEQL    4096
#define ROWS    (N_SEQ * N_WIN * W_WIN)   // 3,131,136 (divisible by 16)
#define TILES   (ROWS / 16)               // 195,696
#define NWIND   (N_SEQ * N_WIN)           // 260,928
#define BM_WPR  (SEQL / 32)               // 128 words per row
#define BM_WORDS (N_SEQ * BM_WPR)         // 8192

// ---------------------------------------------------------------------------
// Kernel 1: fused 2-layer MLP scorer using v_wmma_f32_16x16x32_f16.
// Each wave processes 16 rows per iteration:
//   layer1: (16 x 32) x (32 x 64) via 4 WMMAs (N-tiles of 16), C seeded w/ b1
//   layer2: relu(h) . W2 fused as VALU + 16-lane butterfly reduction
// ---------------------------------------------------------------------------
__global__ __launch_bounds__(256) void mlp_scores_wmma(
    const float* __restrict__ seq,   // (ROWS, 32)
    const float* __restrict__ W1,    // (32, 64)
    const float* __restrict__ b1,    // (64)
    const float* __restrict__ W2,    // (64, 1)
    const float* __restrict__ b2,    // (1)
    float* __restrict__ scores)      // (ROWS)
{
    const int lane = threadIdx.x & 31;
    const int half = lane >> 4;        // 0: K-halves {0..7,16..23}; 1: {8..15,24..31}
    const int sub  = lane & 15;        // row index (A) / column index (B)
    const int wavesPerBlock = blockDim.x >> 5;
    const int wave   = blockIdx.x * wavesPerBlock + (threadIdx.x >> 5);
    const int stride = gridDim.x * wavesPerBlock;

    // ---- Build B tiles (W1 f32 -> f16) once per wave; 4 tiles of N=16 ----
    // Layout (mirror of A): lane holds column N=sub; halves [0..7] = K(half*8..),
    // halves [8..15] = K(16+half*8..).
    v16h btile[4];
    float b1v[4], w2v[4];
    #pragma unroll
    for (int t = 0; t < 4; ++t) {
        const int n = t * 16 + sub;
        #pragma unroll
        for (int i = 0; i < 8; ++i) {
            const int k0 = half * 8 + i;        // 0..7 or 8..15
            const int k1 = 16 + half * 8 + i;   // 16..23 or 24..31
            btile[t][i]     = (_Float16)W1[k0 * HID + n];
            btile[t][8 + i] = (_Float16)W1[k1 * HID + n];
        }
        b1v[t] = b1[n];
        w2v[t] = W2[n];
    }
    const float b2s = b2[0];

    for (int tile = wave; tile < TILES; tile += stride) {
        const int row0 = tile * 16;

        // Prefetch next tile's slice of this lane (speculative, no counter).
        const int ntile = tile + stride;
        if (ntile < TILES)
            __builtin_prefetch(seq + (size_t)(ntile * 16 + sub) * FEAT, 0, 0);

        // ---- Load A tile: row = sub, K-chunks selected by lane half ----
        const float* rowp = seq + (size_t)(row0 + sub) * FEAT;
        const float4* p0 = reinterpret_cast<const float4*>(rowp + half * 8);
        const float4* p1 = reinterpret_cast<const float4*>(rowp + 16 + half * 8);
        const float4 f0 = p0[0], f1 = p0[1];
        const float4 f2 = p1[0], f3 = p1[1];

        v16h a;
        a[0]  = (_Float16)f0.x; a[1]  = (_Float16)f0.y;
        a[2]  = (_Float16)f0.z; a[3]  = (_Float16)f0.w;
        a[4]  = (_Float16)f1.x; a[5]  = (_Float16)f1.y;
        a[6]  = (_Float16)f1.z; a[7]  = (_Float16)f1.w;
        a[8]  = (_Float16)f2.x; a[9]  = (_Float16)f2.y;
        a[10] = (_Float16)f2.z; a[11] = (_Float16)f2.w;
        a[12] = (_Float16)f3.x; a[13] = (_Float16)f3.y;
        a[14] = (_Float16)f3.z; a[15] = (_Float16)f3.w;

        // ---- Layer 1 (4 WMMAs) + fused layer-2 partials ----
        float acc[8] = {0.f, 0.f, 0.f, 0.f, 0.f, 0.f, 0.f, 0.f};
        #pragma unroll
        for (int t = 0; t < 4; ++t) {
            v8f c;
            #pragma unroll
            for (int i = 0; i < 8; ++i) c[i] = b1v[t];
            c = __builtin_amdgcn_wmma_f32_16x16x32_f16(
                    /*neg_a=*/false, a, /*neg_b=*/false, btile[t],
                    /*c_mod=*/(short)0, c, /*reuse_a=*/false, /*reuse_b=*/false);
            const float w2 = w2v[t];
            #pragma unroll
            for (int v = 0; v < 8; ++v)
                acc[v] += fmaxf(c[v], 0.f) * w2;   // relu(h) * W2[n]
        }

        // ---- Reduce over N (16 lanes per half-group) ----
        #pragma unroll
        for (int m = 1; m < 16; m <<= 1) {
            #pragma unroll
            for (int v = 0; v < 8; ++v)
                acc[v] += __shfl_xor(acc[v], m, 16);
        }

        // Group 0 (lanes 0-15) holds rows row0+0..7; group 1 rows row0+8..15.
        if (sub == 0) {
            float4 o0 = make_float4(acc[0] + b2s, acc[1] + b2s,
                                    acc[2] + b2s, acc[3] + b2s);
            float4 o1 = make_float4(acc[4] + b2s, acc[5] + b2s,
                                    acc[6] + b2s, acc[7] + b2s);
            float4* op = reinterpret_cast<float4*>(scores + row0 + half * 8);
            op[0] = o0;
            op[1] = o1;
        }
    }
}

// ---------------------------------------------------------------------------
// Kernel 2: zero the bitmap (8192 words)
// ---------------------------------------------------------------------------
__global__ void bitmap_zero(unsigned* __restrict__ bitmap)
{
    int i = blockIdx.x * blockDim.x + threadIdx.x;
    if (i < BM_WORDS) bitmap[i] = 0u;
}

// ---------------------------------------------------------------------------
// Kernel 3: per-window argmin (first-min, matching jnp.argmin) + bitmap set
// ---------------------------------------------------------------------------
__global__ void argmin_bitmap(const float* __restrict__ scores,
                              unsigned* __restrict__ bitmap)
{
    int idx = blockIdx.x * blockDim.x + threadIdx.x;
    if (idx >= NWIND) return;
    const int nrow = idx / N_WIN;
    const int q    = idx - nrow * N_WIN;

    const float* s = scores + (size_t)idx * W_WIN;
    float best = s[0];
    int   bi   = 0;
    #pragma unroll
    for (int w = 1; w < W_WIN; ++w) {
        const float v = s[w];
        if (v < best) { best = v; bi = w; }
    }
    const int rank = bi + q;   // <= 4087 < 4096
    atomicOr(&bitmap[nrow * BM_WPR + (rank >> 5)], 1u << (rank & 31));
}

// ---------------------------------------------------------------------------
// Kernel 4: popcount-reduce bitmap -> mean density (single block)
// ---------------------------------------------------------------------------
__global__ __launch_bounds__(256) void popcount_mean(
    const unsigned* __restrict__ bitmap, float* __restrict__ out)
{
    __shared__ unsigned sdata[256];
    unsigned t = 0;
    for (int i = threadIdx.x; i < BM_WORDS; i += 256)
        t += __popc(bitmap[i]);
    sdata[threadIdx.x] = t;
    __syncthreads();
    for (int s = 128; s > 0; s >>= 1) {
        if ((int)threadIdx.x < s) sdata[threadIdx.x] += sdata[threadIdx.x + s];
        __syncthreads();
    }
    if (threadIdx.x == 0)
        out[0] = (float)sdata[0] / ((float)SEQL * (float)N_SEQ);
}

// ---------------------------------------------------------------------------
extern "C" void kernel_launch(void* const* d_in, const int* in_sizes, int n_in,
                              void* d_out, int out_size, void* d_ws, size_t ws_size,
                              hipStream_t stream)
{
    const float* seq = (const float*)d_in[0];
    const float* W1  = (const float*)d_in[1];
    const float* b1  = (const float*)d_in[2];
    const float* W2  = (const float*)d_in[3];
    const float* b2  = (const float*)d_in[4];
    float* out = (float*)d_out;

    // Workspace: scores (ROWS floats) then bitmap (BM_WORDS uint32)
    float*    scores = (float*)d_ws;
    unsigned* bitmap = (unsigned*)((char*)d_ws + (size_t)ROWS * sizeof(float));

    // Kernel 1: 256 threads = 8 waves/block; grid-stride over 195,696 tiles.
    const int blocksA = 2048;
    mlp_scores_wmma<<<blocksA, 256, 0, stream>>>(seq, W1, b1, W2, b2, scores);

    bitmap_zero<<<(BM_WORDS + 255) / 256, 256, 0, stream>>>(bitmap);

    argmin_bitmap<<<(NWIND + 255) / 256, 256, 0, stream>>>(scores, bitmap);

    popcount_mean<<<1, 256, 0, stream>>>(bitmap, out);
}
